// O3Transformer_10273561772409
// MI455X (gfx1250) — compile-verified, matrix operand order checked by reading, not run on hardware
//
#include <hip/hip_runtime.h>
#include <hip/hip_bf16.h>
#include <math.h>

// ---------------------------------------------------------------------------
// O(3)-equivariant graph transformer forward pass for MI455X (gfx1250).
// All dense contractions run through a generic wave32 LDS-staged WMMA GEMM
// (v_wmma_f32_16x16x32_f16, f32 accumulate). Irregular graph ops (gather,
// segment softmax, equivariant norm) are VALU kernels with f32 atomics.
// ---------------------------------------------------------------------------

typedef __attribute__((ext_vector_type(16))) _Float16 v16h;
typedef __attribute__((ext_vector_type(8)))  _Float16 v8h;
typedef __attribute__((ext_vector_type(8)))  float    v8f;

#define ACT_NONE 0
#define ACT_SILU 1
#define ACT_ELU  2

static inline int divup(int a, int b) { return (a + b - 1) / b; }

// ---------------------------------------------------------------------------
// Generic GEMM: C[M,N] = act(A[M,K] @ B[K,N] + bias).  A,B,C f32 row-major.
// Block = 256 threads = 8 waves computing a 128x32 C panel; each wave does a
// 16x32 strip = two v_wmma_f32_16x16x32_f16 per k-step, reusing one A frag.
//
// Staging (per 32-wide k-step): A panel 128x32 and B tile 32(k)x32(n) -> LDS
// as f16, two-phase: gather into an explicit register array (so the scheduler
// can batch all global loads instead of serializing through one register),
// then convert+store.  Interior tiles use pointer-stepped unconditional
// loads; edge tiles clamp addresses.  Only A's k-overrun needs zeroing
// (0 * garbage = 0 kills B's), applied as multiply-by-mask so the load stays
// unconditional.  Out-of-range M rows / N cols only produce garbage in C
// entries that are never stored.
//
// Fragment layout (16-bit operands, 16x32):
//   lanes 0-15 : row/col = lane,    K = {0..7, 16..23}
//   lanes 16-31: row/col = lane-16, K = {8..15, 24..31}
// C/D: lane L holds N=L&15, rows M = vgpr + (L>=16 ? 8 : 0).
// LDS rows padded to 40 halves: lane stride 20 words -> conflict-free b128.
// ---------------------------------------------------------------------------
__global__ void k_gemm(const float* __restrict__ A, const float* __restrict__ B,
                       float* __restrict__ C, const float* __restrict__ bias,
                       int M, int N, int K, int lda, int ldb, int ldc, int act)
{
    __shared__ _Float16 Alds[128 * 40];
    __shared__ _Float16 Blds[32 * 40];

    int tid  = threadIdx.x;
    int wave = tid >> 5;
    int lane = tid & 31;
    int hi   = lane >> 4;
    int ln   = lane & 15;
    int rowblk = blockIdx.y * 128;
    int col0   = blockIdx.x * 32;

    // staging coordinates: per-thread fixed column, row stepping by 8
    int sr = tid >> 5;     // 0..7   A row base   / B k base
    int sk = tid & 31;     // 0..31  A k          / B n

    bool fullM = (rowblk + 127) < M;
    bool fullN = (col0 + 31) < N;

    v8f acc0 = {0.f, 0.f, 0.f, 0.f, 0.f, 0.f, 0.f, 0.f};
    v8f acc1 = {0.f, 0.f, 0.f, 0.f, 0.f, 0.f, 0.f, 0.f};

    for (int kk = 0; kk < K; kk += 32) {
        bool fullK = (kk + 31) < K;

        // ---- phase 1: gather A (16 values) and B (4 values) into regs -----
        float va[16];
        float vb[4];
        if (fullM && fullK) {
            const float* ap0 = A + (size_t)(rowblk + sr) * lda + (kk + sk);
#pragma unroll
            for (int j = 0; j < 16; ++j)
                va[j] = ap0[(size_t)(j * 8) * lda];
        } else {
            int gk = kk + sk;
            int ck = gk < K ? gk : (K - 1);
            float km = gk < K ? 1.f : 0.f;        // mask keeps load un-sinkable
#pragma unroll
            for (int j = 0; j < 16; ++j) {
                int gr = rowblk + sr + j * 8;
                int cr = gr < M ? gr : (M - 1);   // rows >= M never stored
                va[j] = A[(size_t)cr * lda + ck] * km;
            }
        }
        if (fullN && fullK) {
            const float* bp0 = B + (size_t)(kk + sr) * ldb + (col0 + sk);
#pragma unroll
            for (int j = 0; j < 4; ++j)
                vb[j] = bp0[(size_t)(j * 8) * ldb];
        } else {
            int gn = col0 + sk;
            int cn = gn < N ? gn : (N - 1);       // cols >= N never stored
#pragma unroll
            for (int j = 0; j < 4; ++j) {
                int gk = kk + sr + j * 8;
                int ck = gk < K ? gk : (K - 1);   // A-side zeroing covers k
                vb[j] = B[(size_t)ck * ldb + cn];
            }
        }
        // ---- phase 2: convert + store to LDS ------------------------------
#pragma unroll
        for (int j = 0; j < 16; ++j)
            Alds[(sr + j * 8) * 40 + sk] = (_Float16)va[j];
#pragma unroll
        for (int j = 0; j < 4; ++j)
            Blds[sk * 40 + (sr + j * 8)] = (_Float16)vb[j];
        __syncthreads();

        // ---- fragments: aligned 16B LDS vector loads ----------------------
        const _Float16* ap  = &Alds[(wave * 16 + ln) * 40 + hi * 8];
        const _Float16* bp0 = &Blds[ln * 40 + hi * 8];
        const _Float16* bp1 = &Blds[(16 + ln) * 40 + hi * 8];
        v8h a0 = *(const v8h*)(ap);
        v8h a1 = *(const v8h*)(ap + 16);
        v8h b0 = *(const v8h*)(bp0);
        v8h b1 = *(const v8h*)(bp0 + 16);
        v8h b2 = *(const v8h*)(bp1);
        v8h b3 = *(const v8h*)(bp1 + 16);
        v16h a  = __builtin_shufflevector(a0, a1, 0, 1, 2, 3, 4, 5, 6, 7,
                                                  8, 9, 10, 11, 12, 13, 14, 15);
        v16h bA = __builtin_shufflevector(b0, b1, 0, 1, 2, 3, 4, 5, 6, 7,
                                                  8, 9, 10, 11, 12, 13, 14, 15);
        v16h bB = __builtin_shufflevector(b2, b3, 0, 1, 2, 3, 4, 5, 6, 7,
                                                  8, 9, 10, 11, 12, 13, 14, 15);
        acc0 = __builtin_amdgcn_wmma_f32_16x16x32_f16(false, a, false, bA,
                                                      (short)0, acc0, false, false);
        acc1 = __builtin_amdgcn_wmma_f32_16x16x32_f16(false, a, false, bB,
                                                      (short)0, acc1, false, false);
        __syncthreads();
    }

    // ---- epilogue ----------------------------------------------------------
    int row0 = rowblk + wave * 16;
#pragma unroll
    for (int t = 0; t < 2; ++t) {
        int bcol = col0 + t * 16 + ln;
        const v8f* accp = t ? &acc1 : &acc0;
        if (bcol < N) {
            float bv = bias ? bias[bcol] : 0.f;
#pragma unroll
            for (int i = 0; i < 8; ++i) {
                int m = row0 + i + hi * 8;
                if (m < M) {
                    float v = (*accp)[i] + bv;
                    if (act == ACT_SILU)      v = v / (1.f + __expf(-v));
                    else if (act == ACT_ELU)  v = (v > 0.f) ? v : (__expf(v) - 1.f);
                    C[(size_t)m * ldc + bcol] = v;
                }
            }
        }
    }
}

// ---------------------------------------------------------------------------
// Elementwise / graph kernels
// ---------------------------------------------------------------------------
__global__ void k_fill(float* p, float v, int n)
{
    int i = blockIdx.x * blockDim.x + threadIdx.x;
    if (i < n) p[i] = v;
}
__global__ void k_fillu(unsigned* p, unsigned v, int n)
{
    int i = blockIdx.x * blockDim.x + threadIdx.x;
    if (i < n) p[i] = v;
}

__global__ void k_embed(const int* __restrict__ x,
                        const float* __restrict__ Ein, const float* __restrict__ Einh,
                        const float* __restrict__ Eat, const float* __restrict__ Eath,
                        float* __restrict__ h0, float* __restrict__ na,
                        float* __restrict__ h1, float* __restrict__ h2, int Nn)
{
    int n = blockIdx.x * blockDim.x + threadIdx.x;
    if (n >= Nn) return;
    int x0 = x[n * 2 + 0], x1 = x[n * 2 + 1];
    for (int c = 0; c < 32; ++c) {
        h0[n * 64 + c]      = Ein[x0 * 32 + c];
        h0[n * 64 + 32 + c] = Einh[x1 * 32 + c];
        na[n * 64 + c]      = Eat[x0 * 32 + c];
        na[n * 64 + 32 + c] = Eath[x1 * 32 + c];
    }
    for (int i = 0; i < 48; ++i) h1[n * 48 + i] = 0.f;
    for (int i = 0; i < 40; ++i) h2[n * 40 + i] = 0.f;
}

__global__ void k_edge_geom(const float* __restrict__ pos,
                            const int* __restrict__ src, const int* __restrict__ dst,
                            const int* __restrict__ eattr, const float* __restrict__ Eedge,
                            float* __restrict__ y1, float* __restrict__ y2,
                            float* __restrict__ rbfea, int E)
{
    int e = blockIdx.x * blockDim.x + threadIdx.x;
    if (e >= E) return;
    int s = src[e], d = dst[e];
    float vx = pos[d * 3 + 0] - pos[s * 3 + 0];
    float vy = pos[d * 3 + 1] - pos[s * 3 + 1];
    float vz = pos[d * 3 + 2] - pos[s * 3 + 2];
    float dd = sqrtf(vx * vx + vy * vy + vz * vz);
    float inv = 1.f / fmaxf(dd, 1e-9f);
    float ux = vx * inv, uy = vy * inv, uz = vz * inv;
    const float s3  = 1.7320508075688772f;   // sqrt(3)
    const float s15 = 3.8729833462074170f;   // sqrt(15)
    const float s5h = 1.1180339887498949f;   // sqrt(5)/2
    y1[e * 3 + 0] = s3 * ux; y1[e * 3 + 1] = s3 * uy; y1[e * 3 + 2] = s3 * uz;
    y2[e * 5 + 0] = s15 * ux * uy;
    y2[e * 5 + 1] = s15 * uy * uz;
    y2[e * 5 + 2] = s5h * (3.f * uz * uz - 1.f);
    y2[e * 5 + 3] = s15 * ux * uz;
    y2[e * 5 + 4] = 0.5f * s15 * (ux * ux - uy * uy);
    const float w = 5.0f / 16.0f;            // MAX_R / N_BASIS
    for (int j = 0; j < 16; ++j) {
        float c = 5.0f * (float)j / 15.0f;   // linspace(0, MAX_R, 16)
        float t = (dd - c) / w;
        rbfea[e * 48 + j] = __expf(-t * t);
    }
    int ea = eattr[e];
    for (int j = 0; j < 32; ++j) rbfea[e * 48 + 16 + j] = Eedge[ea * 32 + j];
}

// Gather source-node features to edges + build tensor-product path inputs.
// Layouts: f1g/crossb = [e][3][16], f2g = [e][5][8], tp2b = [e][5][16].
__global__ void k_gather(const int* __restrict__ src,
                         const float* __restrict__ h0, const float* __restrict__ h1,
                         const float* __restrict__ h2,
                         const float* __restrict__ y1, const float* __restrict__ y2,
                         float* __restrict__ f0g, float* __restrict__ f1g,
                         float* __restrict__ f2g, float* __restrict__ A1,
                         float* __restrict__ A2, float* __restrict__ crossb,
                         float* __restrict__ tp2b, int E, int W0)
{
    int e = blockIdx.x * blockDim.x + threadIdx.x;
    if (e >= E) return;
    int s = src[e];
    for (int c = 0; c < W0; ++c) f0g[(size_t)e * W0 + c] = h0[s * 64 + c];
    for (int i = 0; i < 48; ++i) f1g[e * 48 + i] = h1[s * 48 + i];
    for (int i = 0; i < 40; ++i) f2g[e * 40 + i] = h2[s * 40 + i];
    float b0 = y1[e * 3 + 0], b1 = y1[e * 3 + 1], b2 = y1[e * 3 + 2];
    const float inv_s3 = 0.5773502691896258f; // 1/sqrt(3)
    const float* f1 = f1g + (size_t)e * 48;
    for (int c = 0; c < 16; ++c) {
        float ax = f1[0 * 16 + c], ay = f1[1 * 16 + c], az = f1[2 * 16 + c];
        A1[e * 16 + c] = ax * b0 + ay * b1 + az * b2;
        crossb[e * 48 + 0 * 16 + c] = ay * b2 - az * b1;
        crossb[e * 48 + 1 * 16 + c] = az * b0 - ax * b2;
        crossb[e * 48 + 2 * 16 + c] = ax * b1 - ay * b0;
        tp2b[e * 80 + 0 * 16 + c] = ax * b1 + ay * b0;
        tp2b[e * 80 + 1 * 16 + c] = ay * b2 + az * b1;
        tp2b[e * 80 + 2 * 16 + c] = (2.f * az * b2 - ax * b0 - ay * b1) * inv_s3;
        tp2b[e * 80 + 3 * 16 + c] = ax * b2 + az * b0;
        tp2b[e * 80 + 4 * 16 + c] = ax * b0 - ay * b1;
    }
    float c0 = y2[e * 5 + 0], c1 = y2[e * 5 + 1], c2 = y2[e * 5 + 2];
    float c3 = y2[e * 5 + 3], c4 = y2[e * 5 + 4];
    const float* f2 = f2g + (size_t)e * 40;
    for (int c = 0; c < 8; ++c) {
        A2[e * 8 + c] = f2[0 * 8 + c] * c0 + f2[1 * 8 + c] * c1 + f2[2 * 8 + c] * c2
                      + f2[3 * 8 + c] * c3 + f2[4 * 8 + c] * c4;
    }
}

// dst[r,d] (+)= src[r,d] * u[r/mdiv, uoff+d]
__global__ void k_combine(float* __restrict__ dst, const float* __restrict__ src,
                          const float* __restrict__ u, int R, int D, int ldu,
                          int uoff, int mdiv, int beta)
{
    int t = blockIdx.x * blockDim.x + threadIdx.x;
    if (t >= R * D) return;
    int r = t / D, d = t - r * D;
    float v = src[t] * u[(size_t)(r / mdiv) * ldu + uoff + d];
    dst[t] = beta ? (dst[t] + v) : v;
}

// dst[e][m][d] (+)= q[e,d] * y[e,m] * u[e, uoff+d]   (rank-1 l=0 path)
__global__ void k_combine_outer(float* __restrict__ dst, const float* __restrict__ q,
                                const float* __restrict__ y, const float* __restrict__ u,
                                int E, int MD, int D, int ldu, int uoff, int beta)
{
    int t = blockIdx.x * blockDim.x + threadIdx.x;
    if (t >= E * MD * D) return;
    int d = t % D;
    int m = (t / D) % MD;
    int e = t / (D * MD);
    float v = q[(size_t)e * D + d] * y[(size_t)e * MD + m] * u[(size_t)e * ldu + uoff + d];
    dst[t] = beta ? (dst[t] + v) : v;
}

__global__ void k_score(const int* __restrict__ dst,
                        const float* __restrict__ q0, const float* __restrict__ q1,
                        const float* __restrict__ q2,
                        const float* __restrict__ k0, const float* __restrict__ k1,
                        const float* __restrict__ k2, float* __restrict__ score, int E)
{
    int e = blockIdx.x * blockDim.x + threadIdx.x;
    if (e >= E) return;
    int n = dst[e];
    float s = 0.f;
    for (int c = 0; c < 32; ++c) s += q0[n * 32 + c] * k0[e * 32 + c];
    for (int i = 0; i < 48; ++i) s += q1[n * 48 + i] * k1[e * 48 + i];
    for (int i = 0; i < 40; ++i) s += q2[n * 40 + i] * k2[e * 40 + i];
    score[e] = s * 0.09128709291752768f;   // 1/sqrt(S+3V+5T) = 1/sqrt(120)
}

__device__ inline unsigned enc_f(float x)
{
    unsigned b = __float_as_uint(x);
    return (b & 0x80000000u) ? ~b : (b | 0x80000000u);
}
__device__ inline float dec_f(unsigned u)
{
    unsigned b = (u & 0x80000000u) ? (u & 0x7fffffffu) : ~u;
    return __uint_as_float(b);
}

__global__ void k_segmax(const int* __restrict__ dst, const float* __restrict__ score,
                         unsigned* __restrict__ smax, int E)
{
    int e = blockIdx.x * blockDim.x + threadIdx.x;
    if (e >= E) return;
    atomicMax(&smax[dst[e]], enc_f(score[e]));
}

__global__ void k_expsum(const int* __restrict__ dst, const float* __restrict__ score,
                         const unsigned* __restrict__ smax, float* __restrict__ eexp,
                         float* __restrict__ z, int E)
{
    int e = blockIdx.x * blockDim.x + threadIdx.x;
    if (e >= E) return;
    int n = dst[e];
    float v = __expf(score[e] - dec_f(smax[n]));
    eexp[e] = v;
    atomicAdd(&z[n], v);
}

__global__ void k_aggregate(const int* __restrict__ dst, const float* __restrict__ eexp,
                            const float* __restrict__ z,
                            const float* __restrict__ v0, const float* __restrict__ v1,
                            const float* __restrict__ v2,
                            float* __restrict__ h0, float* __restrict__ h1,
                            float* __restrict__ h2, int E, int o0, int o1, int o2)
{
    int e = blockIdx.x * blockDim.x + threadIdx.x;
    if (e >= E) return;
    int n = dst[e];
    float a = eexp[e] / fmaxf(z[n], 1e-9f);
    for (int c = 0; c < o0; ++c)      atomicAdd(&h0[n * 64 + c], a * v0[e * 32 + c]);
    for (int i = 0; i < 3 * o1; ++i)  atomicAdd(&h1[n * 48 + i], a * v1[e * 48 + i]);
    for (int i = 0; i < 5 * o2; ++i)  atomicAdd(&h2[n * 40 + i], a * v2[e * 40 + i]);
}

__global__ void k_eqnorm(float* __restrict__ h0, float* __restrict__ h1,
                         float* __restrict__ h2,
                         const float* __restrict__ g0, const float* __restrict__ b0,
                         const float* __restrict__ g1, const float* __restrict__ g2,
                         int Nn, int W0, int o1, int o2)
{
    int n = blockIdx.x * blockDim.x + threadIdx.x;
    if (n >= Nn) return;
    float mu = 0.f;
    for (int c = 0; c < W0; ++c) mu += h0[n * 64 + c];
    mu /= (float)W0;
    float var = 0.f;
    for (int c = 0; c < W0; ++c) { float t = h0[n * 64 + c] - mu; var += t * t; }
    var /= (float)W0;
    float inv = rsqrtf(var + 1e-6f);
    for (int c = 0; c < W0; ++c)
        h0[n * 64 + c] = (h0[n * 64 + c] - mu) * inv * g0[c] + b0[c];
    if (o1 > 0) {
        float r = 0.f;
        for (int i = 0; i < 3 * o1; ++i) { float t = h1[n * 48 + i]; r += t * t; }
        float i1 = rsqrtf(r / (float)(3 * o1) + 1e-6f);
        for (int m = 0; m < 3; ++m)
            for (int c = 0; c < o1; ++c) h1[n * 48 + m * 16 + c] *= i1 * g1[c];
    }
    if (o2 > 0) {
        float r = 0.f;
        for (int i = 0; i < 5 * o2; ++i) { float t = h2[n * 40 + i]; r += t * t; }
        float i2 = rsqrtf(r / (float)(5 * o2) + 1e-6f);
        for (int m = 0; m < 5; ++m)
            for (int c = 0; c < o2; ++c) h2[n * 40 + m * 8 + c] *= i2 * g2[c];
    }
}

__global__ void k_ffn_gate(float* __restrict__ mid0, float* __restrict__ mid1,
                           float* __restrict__ mid2, const float* __restrict__ s, int Nn)
{
    int n = blockIdx.x * blockDim.x + threadIdx.x;
    if (n >= Nn) return;
    for (int c = 0; c < 64; ++c) {
        float x = mid0[n * 64 + c];
        mid0[n * 64 + c] = (x / (1.f + __expf(-x))) * s[n * 112 + c];
    }
    for (int c = 0; c < 32; ++c) {
        float sg = 1.f / (1.f + __expf(-s[n * 112 + 64 + c]));
        for (int m = 0; m < 3; ++m) mid1[n * 96 + m * 32 + c] *= sg;
    }
    for (int c = 0; c < 16; ++c) {
        float sg = 1.f / (1.f + __expf(-s[n * 112 + 96 + c]));
        for (int m = 0; m < 5; ++m) mid2[n * 80 + m * 16 + c] *= sg;
    }
}

__global__ void k_head2(const float* __restrict__ t, const float* __restrict__ Wo2,
                        const float* __restrict__ bo2, float* __restrict__ out, int Nn)
{
    int n = blockIdx.x * blockDim.x + threadIdx.x;
    if (n >= Nn) return;
    float s = bo2[0];
    for (int j = 0; j < 96; ++j) s += t[n * 96 + j] * Wo2[j];
    out[n] = s;
}

// ---------------------------------------------------------------------------
// Host helpers
// ---------------------------------------------------------------------------
static void launch_gemm(hipStream_t st, const float* A, const float* B, float* C,
                        const float* bias, int M, int N, int K,
                        int lda, int ldb, int ldc, int act)
{
    dim3 blk(256, 1, 1);
    dim3 grd(divup(N, 32), divup(M, 128), 1);
    k_gemm<<<grd, blk, 0, st>>>(A, B, C, bias, M, N, K, lda, ldb, ldc, act);
}

// Full tensor product: out{0,1,2} = sum_p (path_p @ W_p) * u[:, slice_p]
static void build_tp(hipStream_t st,
                     const float* f0g, const float* f1g, const float* f2g,
                     const float* A1, const float* A2,
                     const float* crossb, const float* tp2b,
                     const float* y1, const float* y2,
                     const float* u, int uw, const float* const W[9],
                     float* o0b, float* o1b, float* o2b, float* gtmp,
                     int E, int i0, int o0, int o1, int o2)
{
    // l=0 output
    launch_gemm(st, f0g, W[0], gtmp, nullptr, E, o0, i0, i0, o0, o0, ACT_NONE);
    k_combine<<<divup(E * o0, 256), 256, 0, st>>>(o0b, gtmp, u, E, o0, uw, 0, 1, 0);
    launch_gemm(st, A1, W[1], gtmp, nullptr, E, o0, 16, 16, o0, o0, ACT_NONE);
    k_combine<<<divup(E * o0, 256), 256, 0, st>>>(o0b, gtmp, u, E, o0, uw, o0, 1, 1);
    launch_gemm(st, A2, W[2], gtmp, nullptr, E, o0, 8, 8, o0, o0, ACT_NONE);
    k_combine<<<divup(E * o0, 256), 256, 0, st>>>(o0b, gtmp, u, E, o0, uw, 2 * o0, 1, 1);
    // l=1 output
    if (o1 > 0) {
        launch_gemm(st, f0g, W[3], gtmp, nullptr, E, o1, i0, i0, o1, o1, ACT_NONE);
        k_combine_outer<<<divup(E * 3 * o1, 256), 256, 0, st>>>(
            o1b, gtmp, y1, u, E, 3, o1, uw, 3 * o0, 0);
        launch_gemm(st, f1g, W[4], gtmp, nullptr, 3 * E, o1, 16, 16, o1, o1, ACT_NONE);
        k_combine<<<divup(3 * E * o1, 256), 256, 0, st>>>(
            o1b, gtmp, u, 3 * E, o1, uw, 3 * o0 + o1, 3, 1);
        launch_gemm(st, crossb, W[5], gtmp, nullptr, 3 * E, o1, 16, 16, o1, o1, ACT_NONE);
        k_combine<<<divup(3 * E * o1, 256), 256, 0, st>>>(
            o1b, gtmp, u, 3 * E, o1, uw, 3 * o0 + 2 * o1, 3, 1);
    }
    // l=2 output
    if (o2 > 0) {
        launch_gemm(st, f0g, W[6], gtmp, nullptr, E, o2, i0, i0, o2, o2, ACT_NONE);
        k_combine_outer<<<divup(E * 5 * o2, 256), 256, 0, st>>>(
            o2b, gtmp, y2, u, E, 5, o2, uw, 3 * o0 + 3 * o1, 0);
        launch_gemm(st, f2g, W[7], gtmp, nullptr, 5 * E, o2, 8, 8, o2, o2, ACT_NONE);
        k_combine<<<divup(5 * E * o2, 256), 256, 0, st>>>(
            o2b, gtmp, u, 5 * E, o2, uw, 3 * o0 + 3 * o1 + o2, 5, 1);
        launch_gemm(st, tp2b, W[8], gtmp, nullptr, 5 * E, o2, 16, 16, o2, o2, ACT_NONE);
        k_combine<<<divup(5 * E * o2, 256), 256, 0, st>>>(
            o2b, gtmp, u, 5 * E, o2, uw, 3 * o0 + 3 * o1 + 2 * o2, 5, 1);
    }
}

// ---------------------------------------------------------------------------
// kernel_launch
// Param leaves assumed in JAX pytree order (sorted dict keys) after the four
// array inputs: Wo1,Wo2,bo1,bo2,emb_attr,emb_attr_h,emb_edge,emb_in,emb_in_h,
// ffn[5]{W1[3],W2[3],Wg1,Wg2,bg1,norm{b0,g0,g1,g2}},
// layers[6]{Wq[3],Wr1,Wrk,Wrv,br1,mixk{w0[3],w1[3],w2[3]},mixv{...},norm{...}}
// ---------------------------------------------------------------------------
extern "C" void kernel_launch(void* const* d_in, const int* in_sizes, int n_in,
                              void* d_out, int out_size, void* d_ws, size_t ws_size,
                              hipStream_t stream)
{
    (void)in_sizes; (void)n_in; (void)out_size; (void)ws_size;
    const int N = 10000, E = 100000;

    const int*   x     = (const int*)d_in[0];
    const float* pos   = (const float*)d_in[1];
    const int*   ei    = (const int*)d_in[2];
    const int*   eattr = (const int*)d_in[3];
    const int*   esrc  = ei;
    const int*   edst  = ei + E;

    const int FFNB = 13;            // first ffn leaf
    const int LYRB = 13 + 5 * 13;   // first layer leaf (=78)
    auto P = [&](int i) -> const float* { return (const float*)d_in[i]; };

    // --- workspace arena ----------------------------------------------------
    float* ws = (float*)d_ws;
    size_t off = 0;
    auto A_ = [&](size_t nf) { float* p = ws + off; off += nf; return p; };
    float* na   = A_((size_t)64 * N);
    float* h0A  = A_((size_t)64 * N);  float* h0B = A_((size_t)64 * N);
    float* h1A  = A_((size_t)48 * N);  float* h1B = A_((size_t)48 * N);
    float* h2A  = A_((size_t)40 * N);  float* h2B = A_((size_t)40 * N);
    float* q0   = A_((size_t)32 * N);
    float* q1   = A_((size_t)48 * N);
    float* q2   = A_((size_t)40 * N);
    float* sbuf = A_((size_t)112 * N);
    float* mid0 = A_((size_t)64 * N);
    float* mid1 = A_((size_t)96 * N);
    float* mid2 = A_((size_t)80 * N);
    float* zbuf = A_((size_t)N);
    unsigned* smax = (unsigned*)A_((size_t)N);
    float* y1    = A_((size_t)3 * E);
    float* y2    = A_((size_t)5 * E);
    float* rbfea = A_((size_t)48 * E);
    float* mbuf  = A_((size_t)64 * E);
    float* ubuf  = A_((size_t)168 * E);
    float* f0g   = A_((size_t)64 * E);
    float* f1g   = A_((size_t)48 * E);
    float* f2g   = A_((size_t)40 * E);
    float* A1b   = A_((size_t)16 * E);
    float* A2b   = A_((size_t)8 * E);
    float* crossb = A_((size_t)48 * E);
    float* tp2b  = A_((size_t)80 * E);
    float* gtmp  = A_((size_t)48 * E);
    float* kv0   = A_((size_t)32 * E);   // k then v (reuse after score)
    float* kv1   = A_((size_t)48 * E);
    float* kv2   = A_((size_t)40 * E);
    float* score = A_((size_t)E);
    float* eexp  = A_((size_t)E);

    // --- embeddings + edge geometry ----------------------------------------
    k_embed<<<divup(N, 128), 128, 0, stream>>>(x, P(11), P(12), P(8), P(9),
                                               h0A, na, h1A, h2A, N);
    k_edge_geom<<<divup(E, 128), 128, 0, stream>>>(pos, esrc, edst, eattr, P(10),
                                                   y1, y2, rbfea, E);

    // --- layers -------------------------------------------------------------
    for (int L = 0; L < 6; ++L) {
        int i0 = (L == 0) ? 64 : 32;
        int o0 = 32, o1 = (L < 5) ? 16 : 0, o2 = (L < 5) ? 8 : 0;
        auto Lw = [&](int o) -> const float* { return (const float*)d_in[LYRB + L * 29 + o]; };
        // leaf offsets: 0-2 Wq, 3 Wr1, 4 Wrk, 5 Wrv, 6 br1,
        //               7-15 mixk, 16-24 mixv, 25 b0, 26 g0, 27 g1, 28 g2

        k_gather<<<divup(E, 128), 128, 0, stream>>>(esrc, h0A, h1A, h2A, y1, y2,
                                                    f0g, f1g, f2g, A1b, A2b,
                                                    crossb, tp2b, E, i0);
        // radial MLP: m = silu(rbf_ea @ Wr1 + br1)
        launch_gemm(stream, rbfea, Lw(3), mbuf, Lw(6), E, 64, 48, 48, 64, 64, ACT_SILU);
        // queries
        launch_gemm(stream, h0A, Lw(0), q0, nullptr, N, 32, i0, 64, 32, 32, ACT_NONE);
        launch_gemm(stream, h1A, Lw(1), q1, nullptr, 3 * N, 16, 16, 16, 16, 16, ACT_NONE);
        launch_gemm(stream, h2A, Lw(2), q2, nullptr, 5 * N, 8, 8, 8, 8, 8, ACT_NONE);
        // keys: uk = m @ Wrk, then tensor product
        launch_gemm(stream, mbuf, Lw(4), ubuf, nullptr, E, 168, 64, 64, 168, 168, ACT_NONE);
        const float* WK[9];
        for (int j = 0; j < 9; ++j) WK[j] = Lw(7 + j);
        build_tp(stream, f0g, f1g, f2g, A1b, A2b, crossb, tp2b, y1, y2,
                 ubuf, 168, WK, kv0, kv1, kv2, gtmp, E, i0, 32, 16, 8);
        // attention scores + segment softmax
        k_score<<<divup(E, 256), 256, 0, stream>>>(edst, q0, q1, q2, kv0, kv1, kv2, score, E);
        k_fillu<<<divup(N, 256), 256, 0, stream>>>(smax, 0u, N);
        k_segmax<<<divup(E, 256), 256, 0, stream>>>(edst, score, smax, E);
        k_fill<<<divup(N, 256), 256, 0, stream>>>(zbuf, 0.f, N);
        k_expsum<<<divup(E, 256), 256, 0, stream>>>(edst, score, smax, eexp, zbuf, E);
        // values: uv = m @ Wrv, tensor product into kv buffers (k now dead)
        int uvw = 3 * (o0 + o1 + o2);
        launch_gemm(stream, mbuf, Lw(5), ubuf, nullptr, E, uvw, 64, 64, uvw, uvw, ACT_NONE);
        const float* WV[9];
        for (int j = 0; j < 9; ++j) WV[j] = Lw(16 + j);
        build_tp(stream, f0g, f1g, f2g, A1b, A2b, crossb, tp2b, y1, y2,
                 ubuf, uvw, WV, kv0, kv1, kv2, gtmp, E, i0, o0, o1, o2);
        // weighted segment-sum into hB
        k_fill<<<divup(64 * N, 256), 256, 0, stream>>>(h0B, 0.f, 64 * N);
        k_fill<<<divup(48 * N, 256), 256, 0, stream>>>(h1B, 0.f, 48 * N);
        k_fill<<<divup(40 * N, 256), 256, 0, stream>>>(h2B, 0.f, 40 * N);
        k_aggregate<<<divup(E, 256), 256, 0, stream>>>(edst, eexp, zbuf, kv0, kv1, kv2,
                                                       h0B, h1B, h2B, E, o0, o1, o2);
        k_eqnorm<<<divup(N, 128), 128, 0, stream>>>(h0B, h1B, h2B,
                                                    Lw(26), Lw(25), Lw(27), Lw(28),
                                                    N, o0, o1, o2);
        if (L < 5) {
            auto Fw = [&](int o) -> const float* { return (const float*)d_in[FFNB + L * 13 + o]; };
            // leaf offsets: 0-2 W1, 3-5 W2, 6 Wg1, 7 Wg2, 8 bg1, 9 b0,10 g0,11 g1,12 g2
            launch_gemm(stream, h0B, Fw(0), mid0, nullptr, N, 64, 32, 64, 64, 64, ACT_NONE);
            launch_gemm(stream, h1B, Fw(1), mid1, nullptr, 3 * N, 32, 16, 16, 32, 32, ACT_NONE);
            launch_gemm(stream, h2B, Fw(2), mid2, nullptr, 5 * N, 16, 8, 8, 16, 16, ACT_NONE);
            launch_gemm(stream, na, Fw(6), gtmp, Fw(8), N, 64, 64, 64, 64, 64, ACT_SILU);
            launch_gemm(stream, gtmp, Fw(7), sbuf, nullptr, N, 112, 64, 64, 112, 112, ACT_NONE);
            k_ffn_gate<<<divup(N, 128), 128, 0, stream>>>(mid0, mid1, mid2, sbuf, N);
            launch_gemm(stream, mid0, Fw(3), h0A, nullptr, N, 32, 64, 64, 32, 64, ACT_NONE);
            launch_gemm(stream, mid1, Fw(4), h1A, nullptr, 3 * N, 16, 32, 32, 16, 16, ACT_NONE);
            launch_gemm(stream, mid2, Fw(5), h2A, nullptr, 5 * N, 8, 16, 16, 8, 8, ACT_NONE);
            k_eqnorm<<<divup(N, 128), 128, 0, stream>>>(h0A, h1A, h2A,
                                                        Fw(10), Fw(9), Fw(11), Fw(12),
                                                        N, 32, 16, 8);
        }
    }

    // --- output head: elu(h0 @ Wo1 + bo1) @ Wo2 + bo2 -----------------------
    launch_gemm(stream, h0B, P(4), sbuf, P(6), N, 96, 32, 64, 96, 96, ACT_ELU);
    k_head2<<<divup(N, 256), 256, 0, stream>>>(sbuf, P(5), P(7), (float*)d_out, N);
}
